// Graph_MERFISH_1331439862502
// MI455X (gfx1250) — matrix operand order repaired and checked by reading, not compiled, not run on hardware
//
#include <hip/hip_runtime.h>

// ---------------- problem constants ----------------
#define NN    50000
#define ERAW  800000
#define EE    850000          // ERAW + NN self loops
#define FIN   500
#define HH    5
#define CC    32
#define HID   160             // HH*CC
#define K1    512             // FIN padded to multiple of 32
#define NEG_GAT 0.2f
#define NEG     0.01f
#define LN_EPS  1e-5f

#define CHUNK_KT 2            // k-steps (of 32) staged into LDS per chunk
#define CHUNK_ELEMS (CHUNK_KT * 10 * 32 * 16)   // bf16 elems per chunk (10240)

typedef __attribute__((ext_vector_type(16))) __bf16 v16bf;
typedef __attribute__((ext_vector_type(8)))  __bf16 v8bf;
typedef __attribute__((ext_vector_type(8)))  float  v8f;
typedef __attribute__((ext_vector_type(4)))  unsigned v4u;
typedef __attribute__((ext_vector_type(8)))  int v8i;
typedef __attribute__((ext_vector_type(4)))  int v4i;
typedef __attribute__((ext_vector_type(4)))  unsigned uint4v;

#if __has_builtin(__builtin_amdgcn_tensor_load_to_lds)
#define HAVE_TDM 1
#else
#define HAVE_TDM 0
#endif

static inline unsigned cdiv_h(size_t a, unsigned b) { return (unsigned)((a + b - 1) / b); }

// ---------------- small helpers ----------------
__device__ __forceinline__ float lrelu(float x, float s) { return x > 0.f ? x : s * x; }

__device__ __forceinline__ unsigned fmap(float f) {
    unsigned u = __float_as_uint(f);
    return (u & 0x80000000u) ? ~u : (u | 0x80000000u);
}
__device__ __forceinline__ float funmap(unsigned u) {
    return __uint_as_float((u & 0x80000000u) ? (u ^ 0x80000000u) : ~u);
}

__device__ __forceinline__ void edge_sd(const long long* __restrict__ ei, int e, int& s, int& d) {
    if (e < ERAW) { s = (int)ei[e]; d = (int)ei[ERAW + e]; }
    else          { s = e - ERAW; d = s; }
}

// ---------------- TDM: 1-D async copy global -> LDS ----------------
// D# per CDNA5 ISA: group0 {count=1, lds_addr, global_addr, type=2},
// group1 {data_size=4B, tensor_dim0=tile_dim0=nelem, dims 1/2 unused}.
__device__ __forceinline__ void tdm_load_1d(const void* gsrc, void* ldst, unsigned nbytes) {
#if HAVE_TDM
    unsigned nelem = nbytes >> 2;                       // 4-byte elements
    unsigned lds_off = (unsigned)(size_t)ldst;          // low 32 bits of flat LDS addr = LDS offset
    unsigned long long ga = (unsigned long long)(size_t)gsrc;
    v4u g0;
    g0[0] = 1u;                                          // count=1 (valid), user mode
    g0[1] = lds_off;                                     // lds_addr (bytes)
    g0[2] = (unsigned)(ga & 0xFFFFFFFFu);                // global_addr[31:0]
    g0[3] = (unsigned)((ga >> 32) & 0x01FFFFFFu)         // global_addr[56:32]
          | (2u << 30);                                  // type=2 ("image")
    v8i g1;
    g1[0] = (int)(2u << 16);                             // data_size=2 -> 4-byte elems
    g1[1] = (int)((nelem & 0xFFFFu) << 16);              // tensor_dim0[15:0]  (bits 63:48)
    g1[2] = (int)(((nelem >> 16) & 0xFFFFu) | (1u << 16)); // tensor_dim0[31:16], tensor_dim1=1
    g1[3] = (int)((nelem & 0xFFFFu) << 16);              // tile_dim0 (bits 127:112)
    g1[4] = 0;                                           // tile_dim1/tile_dim2 unused
    g1[5] = (int)nelem;                                  // tensor_dim0_stride[31:0]
    g1[6] = 0;
    g1[7] = 0;
    v4i z2 = {};
    v4i z3 = {};
#if defined(__clang_major__) && (__clang_major__ >= 23)
    v8i z8 = {};
    __builtin_amdgcn_tensor_load_to_lds(g0, g1, z2, z3, z8, 0);
#else
    __builtin_amdgcn_tensor_load_to_lds(g0, g1, z2, z3, 0);
#endif
#endif
}

// ---------------- prep kernels ----------------
__global__ void zero_u32(unsigned* __restrict__ p, size_t n) {
    size_t i = (size_t)blockIdx.x * blockDim.x + threadIdx.x;
    if (i < n) p[i] = 0u;
}

__global__ void cvt_pad_x(const float* __restrict__ x, __bf16* __restrict__ xb) {
    size_t i = (size_t)blockIdx.x * blockDim.x + threadIdx.x;
    if (i >= (size_t)NN * K1) return;
    int n = (int)(i / K1), k = (int)(i % K1);
    float v = (k < FIN) ? x[(size_t)n * FIN + k] : 0.f;
    xb[i] = (__bf16)v;
}

// W [Krows,HID] f32 -> Bsw bf16, swizzled into WMMA B-fragment order
__global__ void swizzle_w(const float* __restrict__ W, __bf16* __restrict__ Bsw,
                          int Krows, int Kpad) {
    int ksteps = Kpad / 32;
    size_t total = (size_t)ksteps * 10 * 32 * 16;
    size_t i = (size_t)blockIdx.x * blockDim.x + threadIdx.x;
    if (i >= total) return;
    int e    = (int)(i & 15);
    int lane = (int)((i >> 4) & 31);
    int t    = (int)((i >> 9) % 10);
    int kt   = (int)(i / (16 * 32 * 10));
    int k    = kt * 32 + ((e & 8) << 1) + ((lane >> 4) << 3) + (e & 7);
    int col  = t * 16 + (lane & 15);
    float v = (k < Krows) ? W[(size_t)k * HID + col] : 0.f;
    Bsw[i] = (__bf16)v;
}

// ---------------- WMMA GEMM with TDM-staged B panel in LDS ----------------
// out[NN,HID] = A[NN,Kpad] @ B (pre-swizzled). One wave32 per 16-row tile,
// 10 column tiles; B chunks (2 k-steps) DMA'd into double-buffered LDS by the TDM.
__global__ void gemm_wmma(const __bf16* __restrict__ A, const __bf16* __restrict__ Bsw,
                          float* __restrict__ out, int Kpad,
                          const float* __restrict__ bias, int do_leaky) {
    __shared__ __bf16 sB[2][CHUNK_ELEMS];

    int wave = (int)(((size_t)blockIdx.x * blockDim.x + threadIdx.x) >> 5);
    int lane = threadIdx.x & 31;
    bool active = wave < NN / 16;
    int m0 = (active ? wave : 0) * 16;
    int ksteps = Kpad / 32;
    int nch = (ksteps + CHUNK_KT - 1) / CHUNK_KT;

    v8f acc[10];
#pragma unroll
    for (int t = 0; t < 10; ++t) acc[t] = (v8f)(0.f);

    int row = m0 + (lane & 15);
    int k0l = (lane >> 4) << 3; // 0 (lanes 0-15) or 8 (lanes 16-31)
    const __bf16* Arow = A + (size_t)row * Kpad;

#if HAVE_TDM
    if (threadIdx.x < 32) {
        int nk0 = (ksteps < CHUNK_KT) ? ksteps : CHUNK_KT;
        tdm_load_1d(Bsw, &sB[0][0], (unsigned)nk0 * 10 * 32 * 16 * 2);
    }
#endif

    for (int c = 0; c < nch; ++c) {
        int ktbase = c * CHUNK_KT;
        int nk = ksteps - ktbase; if (nk > CHUNK_KT) nk = CHUNK_KT;
#if HAVE_TDM
        if (threadIdx.x < 32) {
            if (c + 1 < nch) {
                int kb2 = (c + 1) * CHUNK_KT;
                int nk2 = ksteps - kb2; if (nk2 > CHUNK_KT) nk2 = CHUNK_KT;
                tdm_load_1d(Bsw + (size_t)kb2 * 10 * 32 * 16, &sB[(c + 1) & 1][0],
                            (unsigned)nk2 * 10 * 32 * 16 * 2);
                __builtin_amdgcn_s_wait_tensorcnt(1);
            } else {
                __builtin_amdgcn_s_wait_tensorcnt(0);
            }
        }
        __syncthreads();
#else
        // fallback: cooperative staged copy (16B per thread per iter)
        {
            const uint4v* src = (const uint4v*)(Bsw + (size_t)ktbase * 10 * 32 * 16);
            uint4v* dst = (uint4v*)&sB[c & 1][0];
            int nvec = nk * 10 * 32 * 16 * 2 / 16;
            for (int i = threadIdx.x; i < nvec; i += blockDim.x) dst[i] = src[i];
        }
        __syncthreads();
#endif
        if (active) {
            for (int ktl = 0; ktl < nk; ++ktl) {
                int kb = (ktbase + ktl) * 32 + k0l;
                v8bf alo = *(const v8bf*)(Arow + kb);        // K = kb .. kb+7
                v8bf ahi = *(const v8bf*)(Arow + kb + 16);   // K = kb+16 .. kb+23
                v16bf af;
#pragma unroll
                for (int e = 0; e < 8; ++e) { af[e] = alo[e]; af[8 + e] = ahi[e]; }

                const __bf16* Bk = &sB[c & 1][(size_t)ktl * 10 * 32 * 16];
#pragma unroll
                for (int t = 0; t < 10; ++t) {
                    v16bf bfrag = *(const v16bf*)(Bk + ((size_t)t * 32 + lane) * 16);
                    acc[t] = __builtin_amdgcn_wmma_f32_16x16x32_bf16(
                        false, af, false, bfrag, (short)0, acc[t], false, false);
                }
            }
        }
        __syncthreads();
    }

    if (!active) return;
    // D layout: lanes 0-15 -> M = r, lanes 16-31 -> M = 8 + r; N = t*16 + (lane&15)
    int rbase = m0 + ((lane >> 4) << 3);
    int cbase = lane & 15;
#pragma unroll
    for (int t = 0; t < 10; ++t) {
#pragma unroll
        for (int r = 0; r < 8; ++r) {
            int rr = rbase + r;
            int cc = t * 16 + cbase;
            float v = acc[t][r];
            if (bias) v += bias[cc];
            if (do_leaky) v = lrelu(v, NEG);
            out[(size_t)rr * HID + cc] = v;
        }
    }
}

// ---------------- attention dot products: a_s/a_d [NN,HH] ----------------
__global__ void att_dots(const float* __restrict__ xl,
                         const float* __restrict__ att_s, const float* __restrict__ att_d,
                         float* __restrict__ as_, float* __restrict__ ad_) {
    int i = (int)((size_t)blockIdx.x * blockDim.x + threadIdx.x);
    if (i >= NN * HH) return;
    int n = i / HH, h = i % HH;
    const float* xr = xl + (size_t)n * HID + h * CC;
    float s = 0.f, d = 0.f;
#pragma unroll
    for (int c = 0; c < CC; ++c) {
        s += xr[c] * att_s[h * CC + c];
        d += xr[c] * att_d[h * CC + c];
    }
    as_[i] = s; ad_[i] = d;
}

// ---------------- edge-phase kernels ----------------
__global__ void edge_logits(const long long* __restrict__ ei,
                            const float* __restrict__ as_, const float* __restrict__ ad_,
                            float* __restrict__ logits, unsigned* __restrict__ lmaxu) {
    size_t i = (size_t)blockIdx.x * blockDim.x + threadIdx.x;
    if (i >= (size_t)EE * HH) return;
    int e = (int)(i / HH), h = (int)(i % HH);
    int s, d; edge_sd(ei, e, s, d);
    float l = as_[(size_t)s * HH + h] + ad_[(size_t)d * HH + h];
    l = lrelu(l, NEG_GAT);
    logits[i] = l;
    atomicMax(&lmaxu[(size_t)d * HH + h], fmap(l));
}

__global__ void edge_alpha(const long long* __restrict__ ei,
                           const float* __restrict__ logits, const unsigned* __restrict__ lmaxu,
                           float* __restrict__ alpha, float* __restrict__ denom) {
    size_t i = (size_t)blockIdx.x * blockDim.x + threadIdx.x;
    if (i >= (size_t)EE * HH) return;
    int e = (int)(i / HH), h = (int)(i % HH);
    int s, d; edge_sd(ei, e, s, d);
    (void)s;
    float a = __expf(logits[i] - funmap(lmaxu[(size_t)d * HH + h]));
    alpha[i] = a;
    atomicAdd(&denom[(size_t)d * HH + h], a);
}

__global__ void edge_norm(const long long* __restrict__ ei,
                          float* __restrict__ alpha, const float* __restrict__ denom) {
    size_t i = (size_t)blockIdx.x * blockDim.x + threadIdx.x;
    if (i >= (size_t)EE * HH) return;
    int e = (int)(i / HH), h = (int)(i % HH);
    int s, d; edge_sd(ei, e, s, d);
    (void)s;
    alpha[i] = alpha[i] / (denom[(size_t)d * HH + h] + 1e-16f);
}

// one wave32 per edge; lane handles channels j*32 + lane (head = j)
__global__ void edge_agg(const long long* __restrict__ ei,
                         const float* __restrict__ alpha, const float* __restrict__ xl,
                         float* __restrict__ agg) {
    int wave = (int)(((size_t)blockIdx.x * blockDim.x + threadIdx.x) >> 5);
    int lane = threadIdx.x & 31;
    if (wave >= EE) return;
    int s, d; edge_sd(ei, wave, s, d);
    const float* xs = xl + (size_t)s * HID;
    float* od = agg + (size_t)d * HID;
#pragma unroll
    for (int j = 0; j < HH; ++j) {
        float a = alpha[(size_t)wave * HH + j];
        atomicAdd(&od[j * CC + lane], xs[j * CC + lane] * a);
    }
}

// ---------------- post-processing ----------------
__global__ void post_ln(const float* __restrict__ agg, const float* __restrict__ b,
                        const float* __restrict__ g, const float* __restrict__ be,
                        __bf16* __restrict__ hb) {
    int node = (int)(((size_t)blockIdx.x * blockDim.x + threadIdx.x) >> 5);
    int lane = threadIdx.x & 31;
    if (node >= NN) return;
    const float* row = agg + (size_t)node * HID;
    float v[HH];
    float sum = 0.f;
#pragma unroll
    for (int j = 0; j < HH; ++j) {
        int c = j * CC + lane;
        float t = row[c] + b[c];
        t = lrelu(t, NEG);
        v[j] = t;
        sum += t;
    }
#pragma unroll
    for (int o = 16; o > 0; o >>= 1) sum += __shfl_xor(sum, o, 32);
    float mu = sum * (1.f / HID);
    float vs = 0.f;
#pragma unroll
    for (int j = 0; j < HH; ++j) { float t = v[j] - mu; vs += t * t; }
#pragma unroll
    for (int o = 16; o > 0; o >>= 1) vs += __shfl_xor(vs, o, 32);
    float inv = rsqrtf(vs * (1.f / HID) + LN_EPS);
#pragma unroll
    for (int j = 0; j < HH; ++j) {
        int c = j * CC + lane;
        float y = (v[j] - mu) * inv * g[c] + be[c];
        hb[(size_t)node * HID + c] = (__bf16)y;
    }
}

__global__ void post_leaky_cvt(const float* __restrict__ agg, const float* __restrict__ b,
                               __bf16* __restrict__ hb) {
    size_t i = (size_t)blockIdx.x * blockDim.x + threadIdx.x;
    if (i >= (size_t)NN * HID) return;
    int c = (int)(i % HID);
    float v = lrelu(agg[i] + b[c], NEG);
    hb[i] = (__bf16)v;
}

// ---------------- host orchestration ----------------
extern "C" void kernel_launch(void* const* d_in, const int* in_sizes, int n_in,
                              void* d_out, int out_size, void* d_ws, size_t ws_size,
                              hipStream_t stream) {
    (void)in_sizes; (void)n_in; (void)out_size; (void)ws_size;

    const float*     x      = (const float*)d_in[0];
    const long long* ei     = (const long long*)d_in[1];
    const float*     W1     = (const float*)d_in[2];
    const float*     att1_s = (const float*)d_in[3];
    const float*     att1_d = (const float*)d_in[4];
    const float*     b1     = (const float*)d_in[5];
    const float*     g1     = (const float*)d_in[6];
    const float*     be1    = (const float*)d_in[7];
    const float*     W2     = (const float*)d_in[8];
    const float*     att2_s = (const float*)d_in[9];
    const float*     att2_d = (const float*)d_in[10];
    const float*     b2     = (const float*)d_in[11];
    const float*     Wm     = (const float*)d_in[12];
    const float*     bm     = (const float*)d_in[13];

    float* out    = (float*)d_out;                 // [NN, HID]
    float* alpha1 = out + (size_t)NN * HID;        // [EE, HH]
    float* alpha2 = alpha1 + (size_t)EE * HH;      // [EE, HH]

    char* ws = (char*)d_ws;
    size_t off = 0;
    auto carve = [&](size_t bytes) { void* p = ws + off; off += (bytes + 255) & ~(size_t)255; return p; };

    __bf16*   xb     = (__bf16*)carve((size_t)NN * K1 * 2);      // reused as xl2 (f32, 32MB<=51.2MB)
    __bf16*   W1sw   = (__bf16*)carve((size_t)(K1 / 32) * 10 * 32 * 16 * 2);
    __bf16*   W2sw   = (__bf16*)carve((size_t)(HID / 32) * 10 * 32 * 16 * 2);
    __bf16*   Wmsw   = (__bf16*)carve((size_t)(HID / 32) * 10 * 32 * 16 * 2);
    float*    xl1    = (float*)carve((size_t)NN * HID * 4);
    float*    as1    = (float*)carve((size_t)NN * HH * 4);
    float*    ad1    = (float*)carve((size_t)NN * HH * 4);
    float*    as2    = (float*)carve((size_t)NN * HH * 4);
    float*    ad2    = (float*)carve((size_t)NN * HH * 4);
    unsigned* lmaxu  = (unsigned*)carve((size_t)NN * HH * 4);
    float*    denom  = (float*)carve((size_t)NN * HH * 4);
    float*    logits = (float*)carve((size_t)EE * HH * 4);
    float*    agg    = (float*)carve((size_t)NN * HID * 4);      // reused both layers
    __bf16*   h1b    = (__bf16*)carve((size_t)NN * HID * 2);     // reused as h2b
    float*    xl2    = (float*)xb;
    __bf16*   h2b    = h1b;

    const unsigned TB = 256;
    const unsigned gemm_blocks = cdiv_h((size_t)(NN / 16) * 32, TB);
    const unsigned eh_blocks   = cdiv_h((size_t)EE * HH, TB);
    const unsigned agg_blocks  = cdiv_h((size_t)EE * 32, TB);
    const unsigned nh_blocks   = cdiv_h((size_t)NN * HH, TB);
    const unsigned nwave_blks  = cdiv_h((size_t)NN * 32, TB);
    const unsigned nhid_blocks = cdiv_h((size_t)NN * HID, TB);

    // ---- prep ----
    cvt_pad_x<<<cdiv_h((size_t)NN * K1, TB), TB, 0, stream>>>(x, xb);
    swizzle_w<<<cdiv_h((size_t)(K1 / 32) * 10 * 32 * 16, TB), TB, 0, stream>>>(W1, W1sw, FIN, K1);
    swizzle_w<<<cdiv_h((size_t)(HID / 32) * 10 * 32 * 16, TB), TB, 0, stream>>>(W2, W2sw, HID, HID);
    swizzle_w<<<cdiv_h((size_t)(HID / 32) * 10 * 32 * 16, TB), TB, 0, stream>>>(Wm, Wmsw, HID, HID);

    // ---- layer 1 ----
    gemm_wmma<<<gemm_blocks, TB, 0, stream>>>(xb, W1sw, xl1, K1, nullptr, 0);
    att_dots<<<nh_blocks, TB, 0, stream>>>(xl1, att1_s, att1_d, as1, ad1);
    zero_u32<<<nh_blocks, TB, 0, stream>>>(lmaxu, (size_t)NN * HH);
    zero_u32<<<nh_blocks, TB, 0, stream>>>((unsigned*)denom, (size_t)NN * HH);
    zero_u32<<<nhid_blocks, TB, 0, stream>>>((unsigned*)agg, (size_t)NN * HID);
    edge_logits<<<eh_blocks, TB, 0, stream>>>(ei, as1, ad1, logits, lmaxu);
    edge_alpha<<<eh_blocks, TB, 0, stream>>>(ei, logits, lmaxu, alpha1, denom);
    edge_norm<<<eh_blocks, TB, 0, stream>>>(ei, alpha1, denom);
    edge_agg<<<agg_blocks, TB, 0, stream>>>(ei, alpha1, xl1, agg);
    post_ln<<<nwave_blks, TB, 0, stream>>>(agg, b1, g1, be1, h1b);

    // ---- layer 2 ----
    gemm_wmma<<<gemm_blocks, TB, 0, stream>>>(h1b, W2sw, xl2, HID, nullptr, 0);
    att_dots<<<nh_blocks, TB, 0, stream>>>(xl2, att2_s, att2_d, as2, ad2);
    zero_u32<<<nh_blocks, TB, 0, stream>>>(lmaxu, (size_t)NN * HH);
    zero_u32<<<nh_blocks, TB, 0, stream>>>((unsigned*)denom, (size_t)NN * HH);
    zero_u32<<<nhid_blocks, TB, 0, stream>>>((unsigned*)agg, (size_t)NN * HID);
    edge_logits<<<eh_blocks, TB, 0, stream>>>(ei, as2, ad2, logits, lmaxu);
    edge_alpha<<<eh_blocks, TB, 0, stream>>>(ei, logits, lmaxu, alpha2, denom);
    edge_norm<<<eh_blocks, TB, 0, stream>>>(ei, alpha2, denom);
    edge_agg<<<agg_blocks, TB, 0, stream>>>(ei, alpha2, xl2, agg);
    post_leaky_cvt<<<nhid_blocks, TB, 0, stream>>>(agg, b2, h2b);

    // ---- final MLP: out = leaky(h2 @ Wm + bm) ----
    gemm_wmma<<<gemm_blocks, TB, 0, stream>>>(h2b, Wmsw, out, HID, bm, 1);
}
